// SimilaritySmoothing_64828236366067
// MI455X (gfx1250) — matrix operand order repaired and checked by reading, not compiled, not run on hardware
//
#include <hip/hip_runtime.h>

// Problem constants (match reference)
#define BB 8
#define LL 2048
#define HH 512
#define PP 256
#define TT 32

typedef __attribute__((ext_vector_type(16))) __bf16        v16bf;
typedef __attribute__((ext_vector_type(2)))  __bf16        v2bf;
typedef __attribute__((ext_vector_type(8)))  float         v8f;
typedef __attribute__((ext_vector_type(8)))  unsigned int  v8u;

// ---- bf16 conversion via native hardware converts ----
static __device__ __forceinline__ unsigned bfpack2(float lo, float hi) {
  v2bf t;
  t.x = (__bf16)lo;
  t.y = (__bf16)hi;
  return __builtin_bit_cast(unsigned, t);
}
static __device__ __forceinline__ unsigned short f2bf(float f) {
  __bf16 h = (__bf16)f;
  return __builtin_bit_cast(unsigned short, h);
}

// ---- WMMA wrapper: D = A(16x32 bf16) * B(32x16 bf16) + C(16x16 f32) ----
static __device__ __forceinline__ v8f wmma_bf16(v8u a, v8u b, v8f c) {
  return __builtin_amdgcn_wmma_f32_16x16x32_bf16(
      false, __builtin_bit_cast(v16bf, a),
      false, __builtin_bit_cast(v16bf, b),
      (short)0, c, false, false);
}

// A-operand K index for VGPR v, lane-half h (ISA 16-bit A 16x32 layout)
static __device__ __forceinline__ int a_koff(int v, int h) {
  return ((v & 4) << 2) + h * 8 + ((v & 3) << 1);
}
// B-operand K index for VGPR v, lane-half h (same K order as A)
static __device__ __forceinline__ int b_koff(int v, int h) {
  return ((v & 4) << 2) + h * 8 + ((v & 3) << 1);
}

// ============================================================
// fp32 -> bf16 bulk convert (pairs)
// ============================================================
__global__ __launch_bounds__(256) void cvt_bf16_kernel(const float* __restrict__ src,
                                                       unsigned* __restrict__ dst,
                                                       int npairs) {
  int i = blockIdx.x * blockDim.x + threadIdx.x;
  if (i < npairs) dst[i] = bfpack2(src[2 * i], src[2 * i + 1]);
}

// ============================================================
// P transpose + convert: Pt[b][p][l] = bf16(P[b][l][p])
// 16x16 LDS tile transpose, grid (L/16, P/16, B), 256 threads
// ============================================================
__global__ __launch_bounds__(256) void pt_kernel(const float* __restrict__ P,
                                                 unsigned short* __restrict__ Pt) {
  __shared__ float tile[16][17];
  const int tx = threadIdx.x & 15;
  const int ty = threadIdx.x >> 4;
  const int l0 = blockIdx.x * 16;
  const int p0 = blockIdx.y * 16;
  const int b  = blockIdx.z;
  tile[ty][tx] = P[((size_t)(b * LL + l0 + ty)) * PP + p0 + tx];
  __syncthreads();
  Pt[((size_t)(b * PP + p0 + ty)) * LL + l0 + tx] = f2bf(tile[tx][ty]);
}

// ============================================================
// group counts / update mask
// ============================================================
__global__ void zero_counts_kernel(int* counts) {
  int i = threadIdx.x;
  if (i < BB * TT) counts[i] = 0;
}
__global__ void count_kernel(const int* __restrict__ questions, int* counts) {
  int idx = blockIdx.x * blockDim.x + threadIdx.x; // [0, B*L)
  int b = idx / LL;
  atomicAdd(&counts[b * TT + questions[idx]], 1);
}
__global__ void update_kernel(const int* __restrict__ questions,
                              const int* __restrict__ mask,
                              const int* __restrict__ counts,
                              int* update) {
  int idx = blockIdx.x * blockDim.x + threadIdx.x;
  int b = idx / LL;
  int q = questions[idx];
  update[idx] = (mask[idx] == 1 && counts[b * TT + q] > 1) ? 1 : 0;
}

// ============================================================
// reciprocal temperature per row: rt = 1/(softplus(h.wt)+0.01)
// one wave per row
// ============================================================
__global__ __launch_bounds__(128) void temps_kernel(const float* __restrict__ X,
                                                    const float* __restrict__ wt,
                                                    float* rtemps) {
  int wave = threadIdx.x >> 5;
  int lane = threadIdx.x & 31;
  int row = blockIdx.x * 4 + wave; // [0, B*L)
  const float* xr = X + (size_t)row * HH;
  float s = 0.f;
  #pragma unroll
  for (int k = 0; k < HH / 32; ++k) s += xr[lane + 32 * k] * wt[lane + 32 * k];
  #pragma unroll
  for (int off = 16; off >= 1; off >>= 1) s += __shfl_xor(s, off, 32);
  if (lane == 0) {
    float sp = (s > 20.f) ? s : log1pf(__expf(s));
    rtemps[row] = 1.f / (sp + 0.01f);
  }
}

// ============================================================
// projection  Y = X @ W^T  (all operands pre-converted bf16)
// grid: (B*L/16, H/64), one wave per 16x64 tile
// ============================================================
__global__ __launch_bounds__(32) void proj_kernel(const unsigned short* __restrict__ Xbf,
                                                  const unsigned short* __restrict__ Wbf,
                                                  unsigned short* __restrict__ Ybf) {
  const int lane = threadIdx.x & 31;
  const int ln16 = lane & 15;
  const int h    = lane >> 4;
  const int i_base = blockIdx.x * 16;
  const int ng     = blockIdx.y; // 64-wide column group

  v8f acc[4];
  #pragma unroll
  for (int t = 0; t < 4; ++t) acc[t] = (v8f)0.f;

  const unsigned short* xr = Xbf + (size_t)(i_base + ln16) * HH;
  for (int kt = 0; kt < HH / 32; ++kt) {
    const int kbase = kt * 32;
    v8u a;
    #pragma unroll
    for (int v = 0; v < 8; ++v)
      a[v] = *(const unsigned*)(xr + kbase + a_koff(v, h));
    #pragma unroll
    for (int nt = 0; nt < 4; ++nt) {
      const unsigned short* wr = Wbf + (size_t)(ng * 64 + nt * 16 + ln16) * HH;
      v8u bb;
      #pragma unroll
      for (int v = 0; v < 8; ++v)
        bb[v] = *(const unsigned*)(wr + kbase + b_koff(v, h));
      acc[nt] = wmma_bf16(a, bb, acc[nt]);
    }
  }
  // C/D layout: VGPR r, lane -> (row = r + 8*h, col = base + ln16)
  #pragma unroll
  for (int nt = 0; nt < 4; ++nt) {
    #pragma unroll
    for (int r = 0; r < 8; ++r) {
      int row = i_base + r + 8 * h;
      int col = ng * 64 + nt * 16 + ln16;
      Ybf[(size_t)row * HH + col] = f2bf(acc[nt][r]);
    }
  }
}

// ============================================================
// flash-style masked attention + P aggregation
// one wave per 16-row i-tile; j in chunks of 32; O = 16x256 f32
// ============================================================
__global__ __launch_bounds__(128) void attn_kernel(
    const unsigned short* __restrict__ Qbf,
    const unsigned short* __restrict__ Kbf,
    const unsigned short* __restrict__ Ptbf,   // [B][P][L] bf16
    const float* __restrict__ rtemps,
    const int* __restrict__ questions,
    const int* __restrict__ update,
    const float* __restrict__ params,
    float* __restrict__ outS) {
  const int wave = threadIdx.x >> 5;
  const int lane = threadIdx.x & 31;
  const int ln16 = lane & 15;
  const int h    = lane >> 4;
  const int tile = blockIdx.x * 4 + wave;  // [0, B*L/16)
  const int b    = tile >> 7;              // tile / (L/16)
  const int i_base = (tile & 127) * 16;
  const float NEG = -1.0e30f;

  __shared__ float eld[4][16 * 32];
  float* els = &eld[wave][0];

  // hoist per-row state (rows r + 8*h of this i-tile)
  int   qi[8]; float rt[8];
  #pragma unroll
  for (int r = 0; r < 8; ++r) {
    int row = b * LL + i_base + r + 8 * h;
    qi[r] = questions[row];
    rt[r] = rtemps[row];
  }

  v8f O[16];
  #pragma unroll
  for (int t = 0; t < 16; ++t) O[t] = (v8f)0.f;
  float mrow[8], srow[8];
  #pragma unroll
  for (int r = 0; r < 8; ++r) { mrow[r] = NEG; srow[r] = 0.f; }

  const unsigned short* Qb  = Qbf + (size_t)(b * LL) * HH;
  const unsigned short* Kb  = Kbf + (size_t)(b * LL) * HH;
  const unsigned short* PtB = Ptbf + (size_t)(b * PP) * LL;

  for (int chunk = 0; chunk < LL / 32; ++chunk) {
    const int j0 = chunk * 32;
    if (chunk + 1 < LL / 32)  // stream next K chunk toward L2
      __builtin_prefetch(Kb + (size_t)(j0 + 32 + lane) * HH, 0, 1);

    // ---- S = Q_i * K_j^T over H (two 16x16 f32 tiles) ----
    v8f s0 = (v8f)0.f, s1 = (v8f)0.f;
    const unsigned short* qrow = Qb + (size_t)(i_base + ln16) * HH;
    const unsigned short* kr0  = Kb + (size_t)(j0 + ln16) * HH;
    const unsigned short* kr1  = kr0 + (size_t)16 * HH;
    for (int kt = 0; kt < HH / 32; ++kt) {
      const int kbase = kt * 32;
      v8u a, b0, b1;
      #pragma unroll
      for (int v = 0; v < 8; ++v) {
        a[v]  = *(const unsigned*)(qrow + kbase + a_koff(v, h));
        int kb = kbase + b_koff(v, h);
        b0[v] = *(const unsigned*)(kr0 + kb);
        b1[v] = *(const unsigned*)(kr1 + kb);
      }
      s0 = wmma_bf16(a, b0, s0);
      s1 = wmma_bf16(a, b1, s1);
    }

    // ---- temperature scale + same-type mask ----
    const int qj0 = questions[b * LL + j0 + ln16];
    const int qj1 = questions[b * LL + j0 + 16 + ln16];
    float f0[8], f1[8];
    #pragma unroll
    for (int r = 0; r < 8; ++r) {
      float a0 = s0[r] * rt[r];
      float a1 = s1[r] * rt[r];
      f0[r] = (qi[r] == qj0) ? a0 : NEG;
      f1[r] = (qi[r] == qj1) ? a1 : NEG;
    }

    // ---- online softmax: row max / exp / row sum ----
    float alpha[8];
    #pragma unroll
    for (int r = 0; r < 8; ++r) {
      float mc = fmaxf(f0[r], f1[r]);
      #pragma unroll
      for (int off = 1; off < 16; off <<= 1) mc = fmaxf(mc, __shfl_xor(mc, off, 32));
      float mn = fmaxf(mrow[r], mc);
      alpha[r] = __expf(mrow[r] - mn);
      mrow[r]  = mn;
      float e0 = __expf(f0[r] - mn);
      float e1 = __expf(f1[r] - mn);
      float cs = e0 + e1;
      #pragma unroll
      for (int off = 1; off < 16; off <<= 1) cs += __shfl_xor(cs, off, 32);
      srow[r] = srow[r] * alpha[r] + cs;
      // stash exp weights to LDS (row-major 16x32 tile)
      els[(r + 8 * h) * 32 + ln16]      = e0;
      els[(r + 8 * h) * 32 + 16 + ln16] = e1;
    }

    __builtin_amdgcn_wave_barrier();
    asm volatile("s_wait_dscnt 0" ::: "memory");

    // ---- read back exp weights in A-fragment layout, bf16 ----
    v8u ae;
    #pragma unroll
    for (int v = 0; v < 8; ++v) {
      int k = a_koff(v, h);
      ae[v] = bfpack2(els[ln16 * 32 + k], els[ln16 * 32 + k + 1]);
    }
    __builtin_amdgcn_wave_barrier();

    // ---- O = O*alpha + attn(16x32) @ P(32x256) ----
    // B[k][n] = P[j0+k][pcol] = Pt[pcol][j0+k]: contiguous u32 pairs
    #pragma unroll
    for (int pt = 0; pt < 16; ++pt) {
      const unsigned short* pr = PtB + (size_t)(pt * 16 + ln16) * LL + j0;
      v8u bp;
      #pragma unroll
      for (int v = 0; v < 8; ++v)
        bp[v] = *(const unsigned*)(pr + b_koff(v, h));
      #pragma unroll
      for (int r = 0; r < 8; ++r) O[pt][r] *= alpha[r];
      O[pt] = wmma_bf16(ae, bp, O[pt]);
    }
  }

  // ---- finalize: normalize, select by update flag, store ----
  float rs[8]; int up[8];
  #pragma unroll
  for (int r = 0; r < 8; ++r) {
    rs[r] = 1.f / srow[r];
    up[r] = update[b * LL + i_base + r + 8 * h];
  }
  #pragma unroll
  for (int pt = 0; pt < 16; ++pt) {
    #pragma unroll
    for (int r = 0; r < 8; ++r) {
      int row = i_base + r + 8 * h;
      int col = pt * 16 + ln16;
      size_t gi = (size_t)(b * LL + row) * PP + col;
      float val = up[r] ? O[pt][r] * rs[r] : params[gi];
      outS[gi] = val;
    }
  }
}

// ============================================================
// Host-side launcher
// ============================================================
extern "C" void kernel_launch(void* const* d_in, const int* in_sizes, int n_in,
                              void* d_out, int out_size, void* d_ws, size_t ws_size,
                              hipStream_t stream) {
  (void)in_sizes; (void)n_in; (void)out_size; (void)ws_size;
  const float* hidden = (const float*)d_in[0];
  const float* params = (const float*)d_in[1];
  const int*   quest  = (const int*)d_in[2];
  const int*   mask   = (const int*)d_in[3];
  const float* WQ     = (const float*)d_in[4];
  const float* WK     = (const float*)d_in[5];
  const float* wt     = (const float*)d_in[6];

  // workspace carve
  char* wsb = (char*)d_ws;
  size_t off = 0;
  const size_t qk_bytes = (size_t)BB * LL * HH * sizeof(unsigned short); // 16 MB
  unsigned short* Qbf = (unsigned short*)(wsb + off); off += qk_bytes;
  unsigned short* Kbf = (unsigned short*)(wsb + off); off += qk_bytes;
  unsigned short* Hbf = (unsigned short*)(wsb + off); off += qk_bytes;
  unsigned short* WQbf = (unsigned short*)(wsb + off); off += (size_t)HH * HH * 2;
  unsigned short* WKbf = (unsigned short*)(wsb + off); off += (size_t)HH * HH * 2;
  unsigned short* Ptbf = (unsigned short*)(wsb + off); off += (size_t)BB * PP * LL * 2;
  float* rtemps = (float*)(wsb + off); off += (size_t)BB * LL * sizeof(float);
  int*   counts = (int*)(wsb + off);   off += 4096;
  int*   update = (int*)(wsb + off);   off += (size_t)BB * LL * sizeof(int);

  // output layout: [hidden copy | smoothed_params]
  hipMemcpyAsync(d_out, hidden, (size_t)BB * LL * HH * sizeof(float),
                 hipMemcpyDeviceToDevice, stream);
  float* outS = (float*)d_out + (size_t)BB * LL * HH;

  // one-time conversions
  {
    int np = BB * LL * HH / 2;   // hidden pairs
    cvt_bf16_kernel<<<(np + 255) / 256, 256, 0, stream>>>(hidden, (unsigned*)Hbf, np);
    int nw = HH * HH / 2;        // weight pairs
    cvt_bf16_kernel<<<(nw + 255) / 256, 256, 0, stream>>>(WQ, (unsigned*)WQbf, nw);
    cvt_bf16_kernel<<<(nw + 255) / 256, 256, 0, stream>>>(WK, (unsigned*)WKbf, nw);
  }
  {
    dim3 tg(LL / 16, PP / 16, BB);
    pt_kernel<<<tg, 256, 0, stream>>>(params, Ptbf);
  }

  zero_counts_kernel<<<1, 256, 0, stream>>>(counts);
  count_kernel<<<(BB * LL) / 256, 256, 0, stream>>>(quest, counts);
  update_kernel<<<(BB * LL) / 256, 256, 0, stream>>>(quest, mask, counts, update);
  temps_kernel<<<(BB * LL) / 4, 128, 0, stream>>>(hidden, wt, rtemps);

  dim3 pg(BB * LL / 16, HH / 64);
  proj_kernel<<<pg, 32, 0, stream>>>(Hbf, WQbf, Qbf);
  proj_kernel<<<pg, 32, 0, stream>>>(Hbf, WKbf, Kbf);

  attn_kernel<<<(BB * LL / 16) / 4, 128, 0, stream>>>(
      Qbf, Kbf, Ptbf, rtemps, quest, update, params, outS);
}
